// MultiHeadAttention_45543833207158
// MI455X (gfx1250) — matrix operand order repaired and checked by reading, not compiled, not run on hardware
//
#include <hip/hip_runtime.h>
#include <stdint.h>

typedef _Float16 f16;
typedef _Float16 v16h __attribute__((ext_vector_type(16)));
typedef _Float16 v8h  __attribute__((ext_vector_type(8)));
typedef _Float16 v4h  __attribute__((ext_vector_type(4)));
typedef float    v8f  __attribute__((ext_vector_type(8)));

#define BATCH 8
#define SEQ   2048
#define EMB   1024
#define NHEAD 16
#define HDIM  64

#define USE_ASYNC_LDS 1

__device__ __forceinline__ v8f wmma_f16(v16h a, v16h b, v8f c) {
  return __builtin_amdgcn_wmma_f32_16x16x32_f16(false, a, false, b, (short)0, c, false, false);
}

__device__ __forceinline__ v16h join8(v8h lo, v8h hi8) {
  v16h r;
#pragma unroll
  for (int i = 0; i < 8; ++i) { r[i] = lo[i]; r[8 + i] = hi8[i]; }
  return r;
}

// ---------------------------------------------------------------- fp32 -> f16
__global__ __launch_bounds__(256) void cvt_f32_f16(const float* __restrict__ src,
                                                   f16* __restrict__ dst, int n4) {
  int i = blockIdx.x * blockDim.x + threadIdx.x;
  if (i < n4) {
    float4 v = ((const float4*)src)[i];
    v4h o;
    o[0] = (f16)v.x; o[1] = (f16)v.y; o[2] = (f16)v.z; o[3] = (f16)v.w;
    *(v4h*)(dst + 4 * (size_t)i) = o;
  }
}

// ---------------------------------------------------------------- flash attention
// 256 threads = 8 waves; wave w owns queries [q0, q0+16); block iterates 32-key
// tiles.  Scores computed TRANSPOSED (S^T = K x Q^T) so each lane owns one query
// column; softmax = per-lane VALU + one shfl_xor(16).  K tile is staged into LDS
// with GLOBAL_LOAD_ASYNC_TO_LDS_B128 (ASYNCcnt); V is staged transposed.
__global__ __launch_bounds__(256) void flash_attn(const f16* __restrict__ qh,
                                                  const f16* __restrict__ kh,
                                                  const f16* __restrict__ vh,
                                                  f16* __restrict__ oh) {
  __shared__ __align__(32) f16 Kt[32 * 64];      // [key][d]
  __shared__ __align__(32) f16 Vt[64 * 32];      // transposed: [d][key]
  __shared__ __align__(32) f16 Pb[8][16 * 32];   // per-wave P: [query][key]

  const int tid  = threadIdx.x;
  const int lane = tid & 31;
  const int wid  = tid >> 5;
  const int nlo  = lane & 15;
  const int hi   = lane >> 4;

  const int blk   = blockIdx.x;
  const int qtile = blk & 15;                    // 16 tiles of 128 queries
  const int h     = (blk >> 4) & (NHEAD - 1);
  const int b     = blk >> 8;
  const int q0    = qtile * 128 + wid * 16;

  const f16* qbase = qh + ((size_t)b * SEQ) * EMB + h * HDIM;
  const f16* kbase = kh + ((size_t)b * SEQ) * EMB + h * HDIM;
  const f16* vbase = vh + ((size_t)b * SEQ) * EMB + h * HDIM;

  // Q^T B-fragments (32x16, contraction dim = d), loaded once.
  const f16* qrow = qbase + (size_t)(q0 + nlo) * EMB;
  const v16h bq0 = join8(*(const v8h*)(qrow +  0 + hi * 8), *(const v8h*)(qrow + 16 + hi * 8));
  const v16h bq1 = join8(*(const v8h*)(qrow + 32 + hi * 8), *(const v8h*)(qrow + 48 + hi * 8));

  v8f acc0 = {}, acc1 = {}, acc2 = {}, acc3 = {};
  float mloc = -1e30f;   // running max (log2 domain) for query nlo
  float lloc = 0.0f;     // running denom for query nlo
  const float kscale = 0.18033688011112042f;  // (1/sqrt(64)) * log2(e)

  // Staging: 256 threads.  K: one 16B async chunk per thread.
  const int krow_s = tid >> 3;          // key row 0..31
  const int kcol_s = (tid & 7) * 8;     // 8-f16 chunk within the 64-wide row
  const uint32_t lds_k = (uint32_t)(uintptr_t)&Kt[krow_s * 64 + kcol_s];

  for (int kt = 0; kt < SEQ; kt += 32) {
    const f16* kg = kbase + (size_t)(kt + krow_s) * EMB + kcol_s;
    const f16* vg = vbase + (size_t)(kt + krow_s) * EMB + kcol_s;
#if USE_ASYNC_LDS
    // direct global -> LDS DMA, tracked by ASYNCcnt
    asm volatile("global_load_async_to_lds_b128 %0, %1, off"
                 :: "v"(lds_k), "v"(kg) : "memory");
#else
    *(v8h*)&Kt[krow_s * 64 + kcol_s] = *(const v8h*)kg;
#endif
    {
      v8h vv = *(const v8h*)vg;
#pragma unroll
      for (int i = 0; i < 8; ++i) Vt[(kcol_s + i) * 32 + krow_s] = vv[i];
    }
    if (kt + 32 < SEQ) {
      __builtin_prefetch(kg + 32 * EMB, 0, 0);
      __builtin_prefetch(vg + 32 * EMB, 0, 0);
    }
#if USE_ASYNC_LDS
    asm volatile("s_wait_asynccnt 0x0" ::: "memory");
#endif
    __syncthreads();

    // ---- S^T tiles: A = K rows (16 keys x 32 d), B = Q^T.  2 WMMAs per tile.
    v8f st0, st1;
    {
      const f16* krow = &Kt[(0 * 16 + nlo) * 64];
      v16h a0 = join8(*(const v8h*)(krow +  0 + hi * 8), *(const v8h*)(krow + 16 + hi * 8));
      v16h a1 = join8(*(const v8h*)(krow + 32 + hi * 8), *(const v8h*)(krow + 48 + hi * 8));
      v8f c = {};
      c   = wmma_f16(a0, bq0, c);
      st0 = wmma_f16(a1, bq1, c);
    }
    {
      const f16* krow = &Kt[(1 * 16 + nlo) * 64];
      v16h a0 = join8(*(const v8h*)(krow +  0 + hi * 8), *(const v8h*)(krow + 16 + hi * 8));
      v16h a1 = join8(*(const v8h*)(krow + 32 + hi * 8), *(const v8h*)(krow + 48 + hi * 8));
      v8f c = {};
      c   = wmma_f16(a0, bq0, c);
      st1 = wmma_f16(a1, bq1, c);
    }

    // ---- per-lane online softmax for query column nlo (log2 domain).
    float x[16];
#pragma unroll
    for (int r = 0; r < 8; ++r) { x[r] = st0[r] * kscale; x[8 + r] = st1[r] * kscale; }
    float mx = x[0];
#pragma unroll
    for (int i = 1; i < 16; ++i) mx = fmaxf(mx, x[i]);
    mx = fmaxf(mx, __shfl_xor(mx, 16, 32));
    const float mnew  = fmaxf(mloc, mx);
    const float alpha = __builtin_amdgcn_exp2f(mloc - mnew);
    float ps = 0.0f;
    v8h p0, p1;
#pragma unroll
    for (int r = 0; r < 8; ++r) {
      float a = __builtin_amdgcn_exp2f(x[r] - mnew);
      float c = __builtin_amdgcn_exp2f(x[8 + r] - mnew);
      ps += a + c;
      p0[r] = (f16)a; p1[r] = (f16)c;
    }
    ps += __shfl_xor(ps, 16, 32);
    lloc = lloc * alpha + ps;
    mloc = mnew;

    // ---- packed P store: lane's 8 probs are consecutive keys of query nlo
    *(v8h*)&Pb[wid][nlo * 32 +  0 + hi * 8] = p0;
    *(v8h*)&Pb[wid][nlo * 32 + 16 + hi * 8] = p1;
    asm volatile("s_wait_dscnt 0" ::: "memory");    // same-wave DS RAW fence

    // ---- broadcast alpha to accumulator rows (row = query r + 8*hi)
#pragma unroll
    for (int r = 0; r < 8; ++r) {
      float ar = __shfl(alpha, r + 8 * hi, 32);
      acc0[r] *= ar; acc1[r] *= ar; acc2[r] *= ar; acc3[r] *= ar;
    }

    // ---- A fragment of P (16 queries x 32 keys)
    const f16* prow = &Pb[wid][nlo * 32];
    v16h ap = *(const v16h*)(prow + hi * 16);

    // ---- O += P x V : B fragments are contiguous rows of transposed Vt
    {
      v16h bv0 = *(const v16h*)&Vt[(0 * 16 + nlo) * 32 + hi * 16];
      v16h bv1 = *(const v16h*)&Vt[(1 * 16 + nlo) * 32 + hi * 16];
      v16h bv2 = *(const v16h*)&Vt[(2 * 16 + nlo) * 32 + hi * 16];
      v16h bv3 = *(const v16h*)&Vt[(3 * 16 + nlo) * 32 + hi * 16];
      acc0 = wmma_f16(ap, bv0, acc0);
      acc1 = wmma_f16(ap, bv1, acc1);
      acc2 = wmma_f16(ap, bv2, acc2);
      acc3 = wmma_f16(ap, bv3, acc3);
    }
    __syncthreads();
  }

  // ---- normalize (broadcast 1/l per row) and store attention output (f16)
  const float linv = __builtin_amdgcn_rcpf(lloc);
#pragma unroll
  for (int r = 0; r < 8; ++r) {
    const float lr = __shfl(linv, r + 8 * hi, 32);
    const int   m  = r + 8 * hi;
    const size_t o = ((size_t)(b * SEQ + q0 + m)) * EMB + h * HDIM + nlo;
    oh[o + 0]  = (f16)(acc0[r] * lr);
    oh[o + 16] = (f16)(acc1[r] * lr);
    oh[o + 32] = (f16)(acc2[r] * lr);
    oh[o + 48] = (f16)(acc3[r] * lr);
  }
}

// ---------------------------------------------------------------- out = A @ W^T + b
// 32x32 output tile per wave: 4 WMMAs per K-step, each A/B fragment used twice.
__global__ __launch_bounds__(256) void out_proj(const f16* __restrict__ A,
                                                const f16* __restrict__ W,
                                                const float* __restrict__ bias,
                                                float* __restrict__ out) {
  const int lane = threadIdx.x & 31;
  const int wid  = threadIdx.x >> 5;
  const int nlo  = lane & 15;
  const int hi   = lane >> 4;

  const int NT    = EMB / 32;                 // 32 N tiles
  const int wtile = blockIdx.x * 8 + wid;
  const int M0    = (wtile / NT) * 32;
  const int N0    = (wtile % NT) * 32;

  const f16* arow0 = A + (size_t)(M0 + nlo) * EMB;
  const f16* arow1 = A + (size_t)(M0 + 16 + nlo) * EMB;
  const f16* wrow0 = W + (size_t)(N0 + nlo) * EMB;
  const f16* wrow1 = W + (size_t)(N0 + 16 + nlo) * EMB;

  v8f acc00 = {}, acc01 = {}, acc10 = {}, acc11 = {};
  for (int k0 = 0; k0 < EMB; k0 += 32) {
    v16h a0 = *(const v16h*)(arow0 + k0 + hi * 16);
    v16h a1 = *(const v16h*)(arow1 + k0 + hi * 16);
    v16h b0 = *(const v16h*)(wrow0 + k0 + hi * 16);
    v16h b1 = *(const v16h*)(wrow1 + k0 + hi * 16);
    acc00 = wmma_f16(a0, b0, acc00);
    acc01 = wmma_f16(a0, b1, acc01);
    acc10 = wmma_f16(a1, b0, acc10);
    acc11 = wmma_f16(a1, b1, acc11);
  }

  const float bv0 = bias[N0 + nlo];
  const float bv1 = bias[N0 + 16 + nlo];
#pragma unroll
  for (int r = 0; r < 8; ++r) {
    const int m0 = M0 + r + 8 * hi;
    const int m1 = m0 + 16;
    out[(size_t)m0 * EMB + N0 + nlo]      = acc00[r] + bv0;
    out[(size_t)m0 * EMB + N0 + 16 + nlo] = acc01[r] + bv1;
    out[(size_t)m1 * EMB + N0 + nlo]      = acc10[r] + bv0;
    out[(size_t)m1 * EMB + N0 + 16 + nlo] = acc11[r] + bv1;
  }
}

// ---------------------------------------------------------------- host launcher
extern "C" void kernel_launch(void* const* d_in, const int* in_sizes, int n_in,
                              void* d_out, int out_size, void* d_ws, size_t ws_size,
                              hipStream_t stream) {
  const float* values  = (const float*)d_in[0];
  const float* keys    = (const float*)d_in[1];
  const float* queries = (const float*)d_in[2];
  const float* W_out   = (const float*)d_in[3];
  const float* b_out   = (const float*)d_in[4];
  float* out = (float*)d_out;

  const size_t NQKV = (size_t)BATCH * SEQ * EMB;   // 16,777,216
  const size_t NW   = (size_t)EMB * EMB;           // 1,048,576

  f16* qh  = (f16*)d_ws;
  f16* kh  = qh + NQKV;
  f16* vh  = kh + NQKV;
  f16* w16 = vh + NQKV;
  f16* oh  = w16 + NW;

  {
    const int t  = 256;
    const int bq = (int)(NQKV / 4 / t);
    const int bw = (int)(NW / 4 / t);
    cvt_f32_f16<<<bq, t, 0, stream>>>(queries, qh, (int)(NQKV / 4));
    cvt_f32_f16<<<bq, t, 0, stream>>>(keys,    kh, (int)(NQKV / 4));
    cvt_f32_f16<<<bq, t, 0, stream>>>(values,  vh, (int)(NQKV / 4));
    cvt_f32_f16<<<bw, t, 0, stream>>>(W_out,   w16, (int)(NW / 4));
  }

  {
    const int blocks = BATCH * NHEAD * (SEQ / 128); // 2048
    flash_attn<<<blocks, 256, 0, stream>>>(qh, kh, vh, oh);
  }

  {
    const int tiles  = (BATCH * SEQ / 32) * (EMB / 32); // 16384
    const int blocks = tiles / 8;                       // 2048
    out_proj<<<blocks, 256, 0, stream>>>(oh, w16, b_out, out);
  }
}
